// SimpleLSTM_24086176596245
// MI455X (gfx1250) — matrix operand order repaired and checked by reading, not compiled, not run on hardware
//
#include <hip/hip_runtime.h>
#include <math.h>

// Problem dimensions (fixed by the reference).
#define B_  64
#define T_  512
#define D_  1024
#define H_  1024
#define N4_ 4096

typedef __attribute__((ext_vector_type(16))) __bf16 v16bf;
typedef __attribute__((ext_vector_type(8)))  float  v8f;

__device__ __forceinline__ float sigf(float x) { return 1.0f / (1.0f + expf(-x)); }

// Build a 16x32 bf16 A-fragment (hi + lo split) for this lane from a row-major
// f32 row. 16-bit A layout (16x32), matching packed B: per lane, element j (0..15) maps to
// k = (lane>>4)*16 + j  => one contiguous 16-float run (two 8-float loads).
__device__ __forceinline__ void build_a_frag(const float* __restrict__ row, int kbase, int hi,
                                             v16bf& ah, v16bf& al) {
  const v8f r1 = *(const v8f*)(row + kbase + hi * 16);
  const v8f r2 = *(const v8f*)(row + kbase + hi * 16 + 8);
#pragma unroll
  for (int j = 0; j < 8; ++j) {
    float f1 = r1[j]; __bf16 h1 = (__bf16)f1;
    ah[j]     = h1;   al[j]     = (__bf16)(f1 - (float)h1);
    float f2 = r2[j]; __bf16 h2 = (__bf16)f2;
    ah[j + 8] = h2;   al[j + 8] = (__bf16)(f2 - (float)h2);
  }
}

__device__ __forceinline__ v8f wmma_bf16(v16bf a, v16bf b, v8f c) {
  return __builtin_amdgcn_wmma_f32_16x16x32_bf16(
      /*neg_a=*/false, a, /*neg_b=*/false, b,
      /*c_mod=*/(short)0, c, /*reuse_a=*/false, /*reuse_b=*/false);
}

// -----------------------------------------------------------------------------
// Pack a KxN (K=1024, N=4096) row-major f32 weight matrix into B-fragment order
// with a bf16 hi/lo split. Output layout: [ntile(256)][ktile(32)][lane(32)][j(16)].
// B layout used here (32x16, contiguous k, same as A): lane = n_local + 16*(k_local>=16); element j of
// the lane covers k_local = 16*(lane>>4) + j, n_local = lane&15.
// -----------------------------------------------------------------------------
__global__ __launch_bounds__(256) void pack_w_kernel(const float* __restrict__ W,
                                                     __bf16* __restrict__ hiOut,
                                                     __bf16* __restrict__ loOut) {
  const int idx = blockIdx.x * 256 + threadIdx.x;   // grid sized exactly
  const int j    = idx & 15;
  const int lane = (idx >> 4) & 31;
  const int rem  = idx >> 9;          // nt*32 + kt
  const int kt   = rem & 31;
  const int nt   = rem >> 5;          // 0..255
  const int k = kt * 32 + ((lane >> 4) << 4) + j;
  const int n = nt * 16 + (lane & 15);
  const float f = W[(size_t)k * N4_ + n];
  const __bf16 h = (__bf16)f;
  hiOut[idx] = h;
  loOut[idx] = (__bf16)(f - (float)h);
}

__global__ __launch_bounds__(256) void init_state_kernel(const float* __restrict__ c0,
                                                         const float* __restrict__ h0,
                                                         float* __restrict__ c_state,
                                                         float* __restrict__ h_state) {
  const int idx = blockIdx.x * 256 + threadIdx.x;
  if (idx < B_ * H_) {
    c_state[idx] = c0[idx];
    h_state[idx] = h0[idx];
  }
}

// -----------------------------------------------------------------------------
// One LSTM time step. Grid: 256 blocks x 512 threads (16 waves).
//   blockIdx.x & 3  -> m-tile (16 batch rows), blockIdx.x >> 2 -> n-tile in H.
//   wave w: gate = w&3 (flax order i,f,g,o), K-slice = w>>2:
//     slice 0/1 -> x_t @ Wx halves, slice 2/3 -> h_prev @ Wh halves.
// 16 waves give ~4 waves/SIMD of latency hiding on the sequential critical
// path. Partial gate tiles are summed in LDS, then the cell update runs.
// -----------------------------------------------------------------------------
__global__ __launch_bounds__(512) void lstm_step_kernel(
    const float*  __restrict__ xs,
    const __bf16* __restrict__ WxPh, const __bf16* __restrict__ WxPl,
    const __bf16* __restrict__ WhPh, const __bf16* __restrict__ WhPl,
    const float*  __restrict__ bias,
    float* __restrict__ c_state,
    const float* __restrict__ h_prev,
    float* __restrict__ h_next,
    float* __restrict__ ys,
    float* __restrict__ cT, float* __restrict__ hT,
    int t, int last)
{
  __shared__ float g4[4][4][16][16];   // [k-slice][gate][m][n], 16 KB

  const int lane = threadIdx.x & 31;
  const int w    = threadIdx.x >> 5;   // 0..15
  const int gate = w & 3;              // i, f, g, o
  const int ks   = w >> 2;             // K-slice 0..3
  const int mt   = blockIdx.x & 3;
  const int ntH  = blockIdx.x >> 2;    // 0..63, n-tile within H
  const int m_base = mt * 16;
  const int hi   = lane >> 4;
  const int lcol = lane & 15;

  // Per-wave source selection (uniform within the wave; EXEC stays all-ones).
  const int ntg = gate * (H_ / 16) + ntH;          // 0..255: n-tile in 4H
  const size_t btil = (size_t)ntg * 32 * 512 + (size_t)lane * 16;

  const float*  arow;
  const __bf16* bh;
  const __bf16* bl;
  int ktBase;
  if (ks < 2) {
    arow   = xs + ((size_t)(m_base + lcol) * T_ + t) * D_;
    bh     = WxPh + btil;
    bl     = WxPl + btil;
    ktBase = ks * 16;
  } else {
    arow   = h_prev + (size_t)(m_base + lcol) * H_;
    bh     = WhPh + btil;
    bl     = WhPl + btil;
    ktBase = (ks - 2) * 16;
  }

  v8f acc = {};
#pragma unroll 1
  for (int i = 0; i < 16; ++i) {
    const int kt = ktBase + i;
    v16bf ah, al;
    build_a_frag(arow, kt * 32, hi, ah, al);
    const v16bf bhv = *(const v16bf*)(bh + (size_t)kt * 512);
    const v16bf blv = *(const v16bf*)(bl + (size_t)kt * 512);
    acc = wmma_bf16(ah, bhv, acc);   // Ahi*Bhi
    acc = wmma_bf16(ah, blv, acc);   // Ahi*Blo
    acc = wmma_bf16(al, bhv, acc);   // Alo*Bhi
  }

  // Stash partial gate tile. C/D layout: VGPR r -> (M = r + 8*hi, N = lcol).
#pragma unroll
  for (int r = 0; r < 8; ++r) {
    g4[ks][gate][r + 8 * hi][lcol] = acc[r];
  }
  __syncthreads();

  // ---- elementwise cell update: 256 elements, threads 0..255 ----
  if (threadIdx.x < 256) {
    const int e = threadIdx.x;
    const int m = e >> 4;
    const int n = e & 15;
    const int ngl = ntH * 16 + n;

    float gi = 0.f, gf = 0.f, gg = 0.f, go = 0.f;
#pragma unroll
    for (int s = 0; s < 4; ++s) {
      gi += g4[s][0][m][n];
      gf += g4[s][1][m][n];
      gg += g4[s][2][m][n];
      go += g4[s][3][m][n];
    }
    gi += bias[0 * H_ + ngl];
    gf += bias[1 * H_ + ngl];
    gg += bias[2 * H_ + ngl];
    go += bias[3 * H_ + ngl];

    const int    bidx = m_base + m;
    const size_t coff = (size_t)bidx * H_ + ngl;

    const float cprev = c_state[coff];
    const float cn = sigf(gf) * cprev + sigf(gi) * tanhf(gg);
    const float hn = sigf(go) * tanhf(cn);

    c_state[coff] = cn;
    h_next[coff]  = hn;
    ys[((size_t)bidx * T_ + t) * H_ + ngl] = hn;
    if (last) { cT[coff] = cn; hT[coff] = hn; }
  }
}

// -----------------------------------------------------------------------------
// Host launcher. d_in order: c0, h0, xs, Wx, Wh, b. d_out: cT | hT | ys (f32).
// ws: packed bf16 hi/lo weights (32 MB) + f32 c state + 2x f32 h buffers.
// -----------------------------------------------------------------------------
extern "C" void kernel_launch(void* const* d_in, const int* in_sizes, int n_in,
                              void* d_out, int out_size, void* d_ws, size_t ws_size,
                              hipStream_t stream) {
  (void)in_sizes; (void)n_in; (void)out_size; (void)ws_size;

  const float* c0 = (const float*)d_in[0];
  const float* h0 = (const float*)d_in[1];
  const float* xs = (const float*)d_in[2];
  const float* Wx = (const float*)d_in[3];
  const float* Wh = (const float*)d_in[4];
  const float* b  = (const float*)d_in[5];

  float* cT = (float*)d_out;
  float* hT = cT + (size_t)B_ * H_;
  float* ys = hT + (size_t)B_ * H_;

  char* p = (char*)d_ws;
  auto carve = [&](size_t bytes) {
    char* r = p;
    p += (bytes + 255) & ~(size_t)255;
    return r;
  };
  const size_t wcount = (size_t)D_ * N4_;          // 4,194,304 per matrix
  __bf16* WxPh = (__bf16*)carve(wcount * 2);
  __bf16* WxPl = (__bf16*)carve(wcount * 2);
  __bf16* WhPh = (__bf16*)carve(wcount * 2);
  __bf16* WhPl = (__bf16*)carve(wcount * 2);
  float* c_state = (float*)carve((size_t)B_ * H_ * 4);
  float* hbuf0   = (float*)carve((size_t)B_ * H_ * 4);
  float* hbuf1   = (float*)carve((size_t)B_ * H_ * 4);

  // Per-call weight packing + state init (deterministic, ws-only writes).
  pack_w_kernel<<<(int)(wcount / 256), 256, 0, stream>>>(Wx, WxPh, WxPl);
  pack_w_kernel<<<(int)(wcount / 256), 256, 0, stream>>>(Wh, WhPh, WhPl);
  init_state_kernel<<<(B_ * H_) / 256, 256, 0, stream>>>(c0, h0, c_state, hbuf0);

  // Sequential scan: one launch per step = device-wide sync; packed Wx/Wh
  // (32 MB) stay resident in the 192 MB L2 across all 512 steps.
  for (int t = 0; t < T_; ++t) {
    float* hp = (t & 1) ? hbuf1 : hbuf0;
    float* hn = (t & 1) ? hbuf0 : hbuf1;
    lstm_step_kernel<<<256, 512, 0, stream>>>(
        xs, WxPh, WxPl, WhPh, WhPl, b,
        c_state, hp, hn, ys, cT, hT, t, (t == T_ - 1) ? 1 : 0);
  }
}
